// Block_74285754352217
// MI455X (gfx1250) — matrix operand order repaired and checked by reading, not compiled, not run on hardware
//
#include <hip/hip_runtime.h>
#include <hip/hip_bf16.h>

typedef __attribute__((ext_vector_type(16))) _Float16 v16h;
typedef __attribute__((ext_vector_type(8)))  _Float16 v8h;
typedef __attribute__((ext_vector_type(8)))  float    v8f;
typedef __attribute__((ext_vector_type(4)))  int      v4i;

union H16 { v16h v; v8h h[2]; };

// ---------------------------------------------------------------------------
// gfx1250 async global->LDS copy (ASYNCcnt-tracked) with sync fallback.
// Builtin signature (probe-derived): param0 = addrspace(1) v4i*, so cast to
// address-space-qualified v4i pointers explicitly (no implicit void* conv).
// ---------------------------------------------------------------------------
#if defined(__gfx1250__) && __has_builtin(__builtin_amdgcn_global_load_async_to_lds_b128)
#define USE_ASYNC 1
#else
#define USE_ASYNC 0
#endif

__device__ __forceinline__ void g2l16(_Float16* l, const _Float16* g) {
#if USE_ASYNC
  __builtin_amdgcn_global_load_async_to_lds_b128(
      (__attribute__((address_space(1))) v4i*)g,
      (__attribute__((address_space(3))) v4i*)l, 0, 0);
#else
  *reinterpret_cast<v8h*>(l) = *reinterpret_cast<const v8h*>(g);
#endif
}

__device__ __forceinline__ void wait_async() {
#if USE_ASYNC
#if __has_builtin(__builtin_amdgcn_s_wait_asynccnt)
  __builtin_amdgcn_s_wait_asynccnt(0);
#else
  asm volatile("s_wait_asynccnt 0x0" ::: "memory");
#endif
#endif
}

// ---------------------------------------------------------------------------
// Fragment loaders (wave32, 16x16x32 f16 WMMA layouts per CDNA5 ISA 7.12.2)
// A (16x32, MxK): lanes 0-15 row=lane, K=kOff+{0..7, 16..23};
//                 lanes 16-31 row=lane-16, K=kOff+{8..15, 24..31}.
// B (32x16, KxN): staged K-contiguous per column (B^T tile in LDS),
//                 lanes 0-15 col=lane, K=kOff+{0..7, 16..23}; lanes 16-31 K=kOff+{8..15, 24..31}.
// ---------------------------------------------------------------------------
__device__ __forceinline__ v16h load_frag_A(const _Float16* smem, int stride,
                                            int rowBase, int kOff) {
  const int lane = threadIdx.x & 31;
  const int row  = rowBase + (lane & 15);
  const int kb   = kOff + ((lane & 16) ? 8 : 0);
  H16 r;
  r.h[0] = *reinterpret_cast<const v8h*>(smem + row * stride + kb);
  r.h[1] = *reinterpret_cast<const v8h*>(smem + row * stride + kb + 16);
  return r.v;
}

__device__ __forceinline__ v16h load_frag_B(const _Float16* smemT, int stride,
                                            int colBase, int kOff) {
  const int lane = threadIdx.x & 31;
  const int col  = colBase + (lane & 15);
  const int kb   = kOff + ((lane & 16) ? 8 : 0);
  H16 r;
  r.h[0] = *reinterpret_cast<const v8h*>(smemT + col * stride + kb);
  r.h[1] = *reinterpret_cast<const v8h*>(smemT + col * stride + kb + 16);
  return r.v;
}

// ---------------------------------------------------------------------------
// Weight convert f32 [K][N] -> f16 transposed [N][K] (one-time, L2-resident)
// ---------------------------------------------------------------------------
__global__ __launch_bounds__(256) void cvt_transpose(const float* __restrict__ W,
                                                     _Float16* __restrict__ Wt,
                                                     int K, int N) {
  const int total = K * N;
  for (int i = blockIdx.x * 256 + threadIdx.x; i < total; i += gridDim.x * 256) {
    const int kk = i / N, nn = i - kk * N;
    Wt[(size_t)nn * K + kk] = (_Float16)W[i];
  }
}

// ---------------------------------------------------------------------------
// LayerNorm over last dim (768), one block per row, writes f16
// ---------------------------------------------------------------------------
__global__ __launch_bounds__(256) void ln_kernel(const float* __restrict__ x,
                                                 const float* __restrict__ g,
                                                 const float* __restrict__ b,
                                                 _Float16* __restrict__ y) {
  __shared__ float red[256];
  const int row = blockIdx.x;
  const float* xr = x + (size_t)row * 768;
  float s1 = 0.f, s2 = 0.f;
  for (int i = threadIdx.x; i < 768; i += 256) {
    float v = xr[i]; s1 += v; s2 += v * v;
  }
  red[threadIdx.x] = s1; __syncthreads();
  for (int o = 128; o > 0; o >>= 1) {
    if (threadIdx.x < o) red[threadIdx.x] += red[threadIdx.x + o];
    __syncthreads();
  }
  const float mean = red[0] * (1.f / 768.f);
  __syncthreads();
  red[threadIdx.x] = s2; __syncthreads();
  for (int o = 128; o > 0; o >>= 1) {
    if (threadIdx.x < o) red[threadIdx.x] += red[threadIdx.x + o];
    __syncthreads();
  }
  const float var = red[0] * (1.f / 768.f) - mean * mean;
  const float rs  = rsqrtf(var + 1e-6f);
  for (int i = threadIdx.x; i < 768; i += 256)
    y[(size_t)row * 768 + i] = (_Float16)((xr[i] - mean) * rs * g[i] + b[i]);
}

// ---------------------------------------------------------------------------
// WMMA GEMM: C[M,N] = A[M,K](f16) * B[K,N](f16, given as B^T[N][K]) + bias.
// Block 256 thr = 8 wave32; tile 128x128; K step 32; wave computes 64x32.
// LDS double-buffered; staging via async global->LDS b128 copies.
// EPI 0: scatter q/k/v; q scaled 1/8; v stored transposed [B,H,Dh,N]
// EPI 1: outF = resid + C              (proj residual, f32)
// EPI 2: outH = gelu(C)               (fc1, f16)
// EPI 3: outF = resid + C              (fc2 residual -> d_out, f32)
// ---------------------------------------------------------------------------
#define SA 40  // LDS stride in halfs (mult of 8 -> 16B aligned rows)

template <int EPI>
__global__ __launch_bounds__(256) void gemm128(
    const _Float16* __restrict__ A, const _Float16* __restrict__ BwT,
    const float* __restrict__ bias, const float* __restrict__ resid,
    float* __restrict__ outF, _Float16* __restrict__ outH,
    _Float16* __restrict__ q16, _Float16* __restrict__ k16,
    _Float16* __restrict__ v16, int M, int N, int K) {
  (void)M;
  __shared__ _Float16 As[2][128 * SA];
  __shared__ _Float16 Bt[2][128 * SA];
  const int tid  = threadIdx.x;
  const int lane = tid & 31, wave = tid >> 5;
  const int m0 = blockIdx.y * 128, n0 = blockIdx.x * 128;
  const int wM = (wave >> 2) * 64;  // 0 / 64
  const int wN = (wave & 3) * 32;   // 0..96

  auto stage = [&](int buf, int k0) {
    for (int c = tid; c < 512; c += 256) {
      const int row = c >> 2, seg = (c & 3) * 8;
      g2l16(&As[buf][row * SA + seg], &A[(size_t)(m0 + row) * K + k0 + seg]);
    }
    for (int c = tid; c < 512; c += 256) {
      const int row = c >> 2, seg = (c & 3) * 8;
      g2l16(&Bt[buf][row * SA + seg], &BwT[(size_t)(n0 + row) * K + k0 + seg]);
    }
  };

  v8f acc[4][2];
#pragma unroll
  for (int i = 0; i < 4; i++)
#pragma unroll
    for (int j = 0; j < 2; j++)
#pragma unroll
      for (int z = 0; z < 8; z++) acc[i][j][z] = 0.f;

  stage(0, 0);
  wait_async();
  __syncthreads();

  const int nt = K >> 5;
  for (int it = 0; it < nt; it++) {
    const int cur = it & 1;
    if (it + 1 < nt) stage(cur ^ 1, (it + 1) * 32);

    v16h bf0 = load_frag_B(&Bt[cur][0], SA, wN, 0);
    v16h bf1 = load_frag_B(&Bt[cur][0], SA, wN + 16, 0);
#pragma unroll
    for (int tm = 0; tm < 4; tm++) {
      v16h af = load_frag_A(&As[cur][0], SA, wM + tm * 16, 0);
      acc[tm][0] = __builtin_amdgcn_wmma_f32_16x16x32_f16(
          false, af, false, bf0, (short)0, acc[tm][0], false, false);
      acc[tm][1] = __builtin_amdgcn_wmma_f32_16x16x32_f16(
          false, af, false, bf1, (short)0, acc[tm][1], false, false);
    }
    wait_async();
    __syncthreads();
  }

  // epilogue; D layout: row = v + (lane&16?8:0), col = lane&15
  const int rOff = (lane & 16) ? 8 : 0;
  const int colL = lane & 15;
#pragma unroll
  for (int tm = 0; tm < 4; tm++)
#pragma unroll
    for (int tn = 0; tn < 2; tn++)
#pragma unroll
      for (int vv = 0; vv < 8; vv++) {
        const int row = m0 + wM + tm * 16 + vv + rOff;
        const int col = n0 + wN + tn * 16 + colL;
        float val = acc[tm][tn][vv] + bias[col];
        if (EPI == 0) {
          const int part = col / 768, j = col % 768;
          const int hh = j >> 6, dh = j & 63;
          const int bb = row >> 11, nn = row & 2047;
          const size_t bhh = (size_t)bb * 12 + hh;
          if (part == 0)
            q16[(bhh * 2048 + nn) * 64 + dh] = (_Float16)(val * 0.125f);
          else if (part == 1)
            k16[(bhh * 2048 + nn) * 64 + dh] = (_Float16)val;
          else  // V transposed: [B,H,Dh,N]
            v16[(bhh * 64 + dh) * 2048 + nn] = (_Float16)val;
        } else if (EPI == 1 || EPI == 3) {
          const size_t idx = (size_t)row * N + col;
          outF[idx] = resid[idx] + val;
        } else {  // EPI == 2: exact GELU
          const float gl = 0.5f * val * (1.f + erff(val * 0.70710678118654752f));
          outH[(size_t)row * N + col] = (_Float16)gl;
        }
      }
}

// ---------------------------------------------------------------------------
// Flash attention: block per (b*h, 64-query chunk); loop 64-key chunks.
// Q,K in [BH,N,Dh]; V pre-transposed [BH,Dh,N] so all staging is b128 copies.
// O accumulators live in registers; rescale feeds WMMA C operand directly.
// ---------------------------------------------------------------------------
#define SQ 72  // half stride (64 + 8)
#define SO 68  // float stride

__global__ __launch_bounds__(256) void attn_fa(
    const _Float16* __restrict__ q, const _Float16* __restrict__ k,
    const _Float16* __restrict__ v, _Float16* __restrict__ out) {
  __shared__ _Float16 Qs[64 * SQ];
  __shared__ _Float16 Ks[2][64 * SQ];  // [key][dh] == B^T for Q K^T
  __shared__ _Float16 Vt[2][64 * SQ];  // [dh][key] == B^T for P V
  __shared__ _Float16 Pb[64 * SQ];
  __shared__ float Sf[64 * SO];
  __shared__ float mrow[64], lrow[64], srow[64];

  const int tid = threadIdx.x, lane = tid & 31, wave = tid >> 5;
  const int bh = blockIdx.y;  // 0..23 = b*12 + h
  const int qb = blockIdx.x * 64;
  const size_t kqbase = (size_t)bh * 2048 * 64;  // Q,K: [bh][n][dh]
  const size_t vbase  = (size_t)bh * 64 * 2048;  // V:   [bh][dh][n]

  auto stageKV = [&](int buf, int kt) {
    for (int c = tid; c < 512; c += 256) {
      const int row = c >> 3, seg = (c & 7) * 8;
      g2l16(&Ks[buf][row * SQ + seg],
            &k[kqbase + (size_t)(kt + row) * 64 + seg]);
    }
    for (int c = tid; c < 512; c += 256) {
      const int dh = c >> 3, seg = (c & 7) * 8;
      g2l16(&Vt[buf][dh * SQ + seg], &v[vbase + (size_t)dh * 2048 + kt + seg]);
    }
  };

  for (int c = tid; c < 512; c += 256) {
    const int row = c >> 3, seg = (c & 7) * 8;
    g2l16(&Qs[row * SQ + seg], &q[kqbase + (size_t)(qb + row) * 64 + seg]);
  }
  stageKV(0, 0);
  if (tid < 64) { mrow[tid] = -3.0e38f; lrow[tid] = 0.f; }
  wait_async();
  __syncthreads();

  const int tm  = wave & 3;         // 16-row tile of S / O
  const int tn0 = (wave >> 2) * 2;  // two 16-col tiles
  const int rOff = (lane & 16) ? 8 : 0;
  const int colL = lane & 15;

  v8f accO[2];
#pragma unroll
  for (int t = 0; t < 2; t++)
#pragma unroll
    for (int z = 0; z < 8; z++) accO[t][z] = 0.f;

  for (int it = 0; it < 32; it++) {
    const int cur = it & 1;
    const int kt = it * 64;
    if (it + 1 < 32) stageKV(cur ^ 1, kt + 64);

    // S = Q K^T (scale folded into Q)
#pragma unroll
    for (int t = 0; t < 2; t++) {
      const int tn = tn0 + t;
      v8f acc;
#pragma unroll
      for (int z = 0; z < 8; z++) acc[z] = 0.f;
#pragma unroll
      for (int kk = 0; kk < 64; kk += 32) {
        v16h a = load_frag_A(Qs, SQ, tm * 16, kk);
        v16h b = load_frag_B(&Ks[cur][0], SQ, tn * 16, kk);
        acc = __builtin_amdgcn_wmma_f32_16x16x32_f16(false, a, false, b,
                                                     (short)0, acc, false, false);
      }
#pragma unroll
      for (int vv = 0; vv < 8; vv++)
        Sf[(tm * 16 + vv + rOff) * SO + tn * 16 + colL] = acc[vv];
    }
    __syncthreads();

    // online softmax, one thread per row
    if (tid < 64) {
      const float m = mrow[tid];
      float rmax = -3.0e38f;
      for (int c2 = 0; c2 < 64; c2++) rmax = fmaxf(rmax, Sf[tid * SO + c2]);
      const float mn = fmaxf(m, rmax);
      const float r = __expf(m - mn);
      float s = 0.f;
      for (int c2 = 0; c2 < 64; c2++) {
        const float p = __expf(Sf[tid * SO + c2] - mn);
        Pb[tid * SQ + c2] = (_Float16)p;
        s += p;
      }
      mrow[tid] = mn; lrow[tid] = lrow[tid] * r + s; srow[tid] = r;
    }
    __syncthreads();

    // O = O*rescale + P V   (rescaled accumulator feeds WMMA C operand)
#pragma unroll
    for (int t = 0; t < 2; t++) {
      const int tn = tn0 + t;
#pragma unroll
      for (int vv = 0; vv < 8; vv++)
        accO[t][vv] *= srow[tm * 16 + vv + rOff];
#pragma unroll
      for (int kk = 0; kk < 64; kk += 32) {
        v16h a = load_frag_A(Pb, SQ, tm * 16, kk);
        v16h b = load_frag_B(&Vt[cur][0], SQ, tn * 16, kk);
        accO[t] = __builtin_amdgcn_wmma_f32_16x16x32_f16(
            false, a, false, b, (short)0, accO[t], false, false);
      }
    }
    wait_async();
    __syncthreads();
  }

  // normalize and write attn_out f16 in [B,N,D] layout
  const int bb = bh / 12, hh = bh % 12;
#pragma unroll
  for (int t = 0; t < 2; t++)
#pragma unroll
    for (int vv = 0; vv < 8; vv++) {
      const int row = tm * 16 + vv + rOff;
      const int col = (tn0 + t) * 16 + colL;
      const float val = accO[t][vv] / lrow[row];
      out[((size_t)(bb * 2048 + qb + row)) * 768 + hh * 64 + col] = (_Float16)val;
    }
}

// ---------------------------------------------------------------------------
// Host-side orchestration
// ---------------------------------------------------------------------------
extern "C" void kernel_launch(void* const* d_in, const int* in_sizes, int n_in,
                              void* d_out, int out_size, void* d_ws, size_t ws_size,
                              hipStream_t stream) {
  (void)in_sizes; (void)n_in; (void)out_size; (void)ws_size;
  const float* x     = (const float*)d_in[0];
  const float* Wqkv  = (const float*)d_in[1];
  const float* bqkv  = (const float*)d_in[2];
  const float* Wproj = (const float*)d_in[3];
  const float* bproj = (const float*)d_in[4];
  const float* Wfc1  = (const float*)d_in[5];
  const float* bfc1  = (const float*)d_in[6];
  const float* Wfc2  = (const float*)d_in[7];
  const float* bfc2  = (const float*)d_in[8];
  const float* ln1g  = (const float*)d_in[9];
  const float* ln1b  = (const float*)d_in[10];
  const float* ln2g  = (const float*)d_in[11];
  const float* ln2b  = (const float*)d_in[12];
  float* out = (float*)d_out;

  const size_t M = 4096;  // B*N = 2*2048
  char* w = (char*)d_ws;
  auto alloc = [&](size_t bytes) -> char* {
    char* p = w; w += (bytes + 255) & ~(size_t)255; return p;
  };
  _Float16* ln1h   = (_Float16*)alloc(M * 768 * 2);
  _Float16* WqT    = (_Float16*)alloc((size_t)768 * 2304 * 2);  // [2304][768]
  _Float16* WpT    = (_Float16*)alloc((size_t)768 * 768 * 2);   // [768][768]
  _Float16* W1T    = (_Float16*)alloc((size_t)768 * 3072 * 2);  // [3072][768]
  _Float16* W2T    = (_Float16*)alloc((size_t)3072 * 768 * 2);  // [768][3072]
  _Float16* q16    = (_Float16*)alloc(M * 768 * 2);
  _Float16* k16    = (_Float16*)alloc(M * 768 * 2);
  _Float16* vv16   = (_Float16*)alloc(M * 768 * 2);
  _Float16* attn16 = (_Float16*)alloc(M * 768 * 2);
  float*    x2     = (float*)   alloc(M * 768 * 4);
  _Float16* h16    = (_Float16*)alloc(M * 768 * 2);
  _Float16* g16    = (_Float16*)alloc(M * 3072 * 2);

  // weight conversion f32 -> f16 transposed
  cvt_transpose<<<2048, 256, 0, stream>>>(Wqkv,  WqT, 768, 2304);
  cvt_transpose<<<1024, 256, 0, stream>>>(Wproj, WpT, 768, 768);
  cvt_transpose<<<2048, 256, 0, stream>>>(Wfc1,  W1T, 768, 3072);
  cvt_transpose<<<2048, 256, 0, stream>>>(Wfc2,  W2T, 3072, 768);

  // LN1 -> f16
  ln_kernel<<<4096, 256, 0, stream>>>(x, ln1g, ln1b, ln1h);

  // QKV GEMM (4096 x 2304 x 768), scatter to q/k [B,H,N,Dh], v [B,H,Dh,N]
  gemm128<0><<<dim3(18, 32), 256, 0, stream>>>(
      ln1h, WqT, bqkv, nullptr, nullptr, nullptr, q16, k16, vv16,
      4096, 2304, 768);

  // Attention: 32 query chunks x 24 (b,h)
  attn_fa<<<dim3(32, 24), 256, 0, stream>>>(q16, k16, vv16, attn16);

  // Proj GEMM + residual -> x2 (f32)
  gemm128<1><<<dim3(6, 32), 256, 0, stream>>>(
      attn16, WpT, bproj, x, x2, nullptr, nullptr, nullptr, nullptr,
      4096, 768, 768);

  // LN2 -> f16
  ln_kernel<<<4096, 256, 0, stream>>>(x2, ln2g, ln2b, h16);

  // FC1 GEMM + GELU -> g16 (f16)
  gemm128<2><<<dim3(24, 32), 256, 0, stream>>>(
      h16, W1T, bfc1, nullptr, nullptr, g16, nullptr, nullptr, nullptr,
      4096, 3072, 768);

  // FC2 GEMM + residual -> out (f32)
  gemm128<3><<<dim3(6, 32), 256, 0, stream>>>(
      g16, W2T, bfc2, x2, out, nullptr, nullptr, nullptr, nullptr,
      4096, 768, 3072);
}
